// BaseModel_46016279609980
// MI455X (gfx1250) — matrix operand order repaired and checked by reading, not compiled, not run on hardware
//
#include <hip/hip_runtime.h>
#include <hip/hip_bf16.h>
#include <math.h>

typedef __bf16 bf16;
typedef __attribute__((ext_vector_type(16))) __bf16 v16bf;
typedef __attribute__((ext_vector_type(8)))  float  v8f;
typedef __attribute__((ext_vector_type(4)))  unsigned int u32x4;
typedef __attribute__((ext_vector_type(8)))  int          i32x8;
typedef __attribute__((ext_vector_type(4)))  int          i32x4;

#if defined(__has_builtin)
#  if __has_builtin(__builtin_amdgcn_tensor_load_to_lds) && \
      __has_builtin(__builtin_amdgcn_s_wait_tensorcnt)
#    define HAVE_TDM 1
#  endif
#endif
#ifndef HAVE_TDM
#  define HAVE_TDM 0
#endif

#define NB    64        // B: batch == num_layers == seq len inside multilayer LSTM
#define LB    50
#define NOUT  3
#define FN    20
#define UU    128
#define GG    512       // 4*U

// Whh staged row-major with TDM padding: 128 bf16 (64 dwords) + 4 pad dwords per row
#define ROW_BF16     136                    // 272 B row stride (16B aligned, bank-rotated)
// ---- dynamic LDS layout for the multilayer LSTM mega-kernel ----
#define LDS_XW_OFF   0                      // float [64][512]      = 131072 B
#define LDS_WHH_OFF  131072                 // bf16  [512][ROW_BF16]= 139264 B (padded rows)
#define LDS_A_OFF    270336                 // bf16  [64][256 max]  =  32768 B (layer input)
#define LDS_H_OFF    303104                 // float [128]
#define LDS_C_OFF    303616                 // float [128]
#define LDS_G_OFF    304128                 // float [512]
#define LDS_TOTAL    306176

__device__ __forceinline__ float sigmoidf_(float x) { return 1.0f / (1.0f + expf(-x)); }

// ---------------- weight prep ----------------
__global__ void cvt_f32_bf16(bf16* __restrict__ dst, const float* __restrict__ src, int n) {
  for (int i = blockIdx.x * blockDim.x + threadIdx.x; i < n; i += gridDim.x * blockDim.x)
    dst[i] = (bf16)src[i];
}

// enc_Wih0 (512x20) -> bf16 padded (512x32), zeros in K=20..31
__global__ void pad_wih0(bf16* __restrict__ dst, const float* __restrict__ src) {
  for (int i = blockIdx.x * blockDim.x + threadIdx.x; i < 512 * 32; i += gridDim.x * blockDim.x) {
    int r = i >> 5, k = i & 31;
    dst[i] = (k < FN) ? (bf16)src[r * FN + k] : (bf16)0.0f;
  }
}

__global__ void zero_f32(float* __restrict__ p, int n) {
  for (int i = blockIdx.x * blockDim.x + threadIdx.x; i < n; i += gridDim.x * blockDim.x)
    p[i] = 0.0f;
}

// ---------------- encoder per-step attention ----------------
__global__ void enc_attn(const float* __restrict__ x, const float* __restrict__ attnW,
                         const float* __restrict__ attnB, const float* __restrict__ hS,
                         bf16* __restrict__ Abuf, int s) {
  int b = blockIdx.x, lane = threadIdx.x;          // block = 1 wave32
  __shared__ float xs[FN], en[FN], hb[UU];
  if (lane < FN) xs[lane] = x[b * (LB * FN) + s * FN + lane];
  for (int k = lane; k < UU; k += 32) hb[k] = hS[b * UU + k];
  __syncthreads();
  if (lane < FN) {
    const float* w = attnW + (size_t)s * FN * (FN + UU) + lane * (FN + UU);
    float acc = attnB[s * FN + lane];
    for (int k = 0; k < FN; ++k) acc = fmaf(w[k], xs[k], acc);
    for (int k = 0; k < UU; ++k) acc = fmaf(w[FN + k], hb[k], acc);
    en[lane] = tanhf(acc);
  }
  __syncthreads();
  if (lane < FN) {
    float mx = -1e30f;
    for (int k = 0; k < FN; ++k) mx = fmaxf(mx, en[k]);
    float se = 0.f;
    for (int k = 0; k < FN; ++k) se += expf(en[k] - mx);
    Abuf[b * 32 + lane] = (bf16)((expf(en[lane] - mx) / se) * xs[lane]);
  } else {
    Abuf[b * 32 + lane] = (bf16)0.0f;              // K padding
  }
}

// ---------------- 64-layer LSTM over 64 steps, single workgroup ----------------
__global__ __launch_bounds__(512) void multilayer_lstm(
    const bf16* __restrict__ A0, int K0,               // layer-0 input (64 x K0)
    const bf16* __restrict__ Wih0,                     // 512 x K0
    const bf16* __restrict__ WihR,                     // 63 x 512 x 128
    const bf16* __restrict__ WhhA,                     // 64 x 512 x 128
    const float* __restrict__ bih0, const float* __restrict__ bhh0,
    const float* __restrict__ bihR, const float* __restrict__ bhhR,
    float* __restrict__ hS, float* __restrict__ cS,    // 64 x 128 per-layer states
    float* __restrict__ hsOut)                         // last-layer hs (64x128) or null
{
  extern __shared__ char smem[];
  float* ldsXW  = (float*)(smem + LDS_XW_OFF);
  bf16*  ldsWhB = (bf16*) (smem + LDS_WHH_OFF);   // [gate j][ROW_BF16] padded rows
  bf16*  ldsA   = (bf16*) (smem + LDS_A_OFF);
  float* ldsH   = (float*)(smem + LDS_H_OFF);
  float* ldsC   = (float*)(smem + LDS_C_OFF);
  float* ldsG   = (float*)(smem + LDS_G_OFF);

  const int tid  = threadIdx.x;
  const int lane = tid & 31;
  const int wave = tid >> 5;                      // 16 waves
  const int m    = lane & 15;                     // A row / B col within tile
  const int kh   = (lane >> 4) * 8;               // A K-half select
  const int kh2  = (lane >> 4) * 16;              // unused; B also uses kh

  for (int l = 0; l < NB; ++l) {
    const int   Kin = (l == 0) ? K0 : UU;
    const bf16* Wih = (l == 0) ? Wih0 : WihR + (size_t)(l - 1) * GG * UU;
    const bf16* Whh = WhhA + (size_t)l * GG * UU;
    const float* bi = (l == 0) ? bih0 : bihR + (l - 1) * GG;
    const float* bh = (l == 0) ? bhh0 : bhhR + (l - 1) * GG;

    __syncthreads();  // prev layer fully done: ldsA holds input, ldsXW/ldsWhB free

#if HAVE_TDM
    // Tensor Data Mover: DMA Whh (512x128 bf16) row-major into LDS with
    // 4-dword padding every 64 dwords (row stride 272B). Overlaps the GEMM.
    if (wave == 0) {
      unsigned long long ga = (unsigned long long)(uintptr_t)Whh;
      u32x4 g0;
      g0[0] = 1u;                                            // count=1, no gather
      g0[1] = (unsigned int)LDS_WHH_OFF;                     // lds_addr
      g0[2] = (unsigned int)(ga & 0xFFFFFFFFu);              // global_addr lo
      g0[3] = (unsigned int)((ga >> 32) & 0x1FFFFFFu) | (2u << 30);  // hi | type=2
      i32x8 g1;
      g1[0] = (int)((1u << 16) | (1u << 20) | (5u << 22) | (3u << 25));
                       // data_size=2B | pad_enable | pad_interval=64dw | pad_amount=4dw
      g1[1] = (int)(128u << 16);   // tensor_dim0 = 128 (lo16)
      g1[2] = (int)(512u << 16);   // tensor_dim1 = 512 (lo16)
      g1[3] = (int)(128u << 16);   // tile_dim0 = 128
      g1[4] = (int)512u;           // tile_dim1 = 512, tile_dim2 = 0
      g1[5] = (int)128u;           // tensor_dim0_stride = 128
      g1[6] = 0; g1[7] = 0;
      i32x4 z4 = {0, 0, 0, 0};
#if __clang_major__ >= 23
      i32x8 z8 = {0, 0, 0, 0, 0, 0, 0, 0};
      __builtin_amdgcn_tensor_load_to_lds(g0, g1, z4, z4, z8, 0);
#else
      __builtin_amdgcn_tensor_load_to_lds(g0, g1, z4, z4, 0);
#endif
    }
#else
    // fallback: manual b128 staging into the same padded row-major layout
    {
      const uint4* W4 = (const uint4*)Whh;        // 8 bf16 per uint4, row-major [j][k]
      for (int q = tid; q < GG * UU / 8; q += 512) {
        int j = q >> 4, ch = q & 15;
        *(uint4*)(ldsWhB + (size_t)j * ROW_BF16 + ch * 8) = W4[q];
      }
    }
#endif

    // layer-0 input: stage from global into LDS (b128 copies)
    if (l == 0) {
      const uint4* A4 = (const uint4*)A0;
      for (int q = tid; q < NB * Kin / 8; q += 512) ((uint4*)ldsA)[q] = A4[q];
    }
    __syncthreads();  // ldsA ready for GEMM

    // WMMA GEMM: XW(64x512) = A(64xKin) x Wih^T(Kinx512); 128 tiles, 8 per wave
    for (int tile = wave; tile < 128; tile += 16) {
      int tm = tile >> 5, tn = tile & 31;
      const bf16* arow = ldsA + (tm * 16 + m) * Kin;
      const bf16* brow = Wih + (size_t)(tn * 16 + m) * Kin;
      v8f acc = {};
      for (int kc = 0; kc < Kin; kc += 32) {
        union { v16bf v; uint4 q[2]; } ua, ub;
        ua.q[0] = *(const uint4*)(arow + kc + kh);        // ds_load_b128
        ua.q[1] = *(const uint4*)(arow + kc + 16 + kh);
        ub.q[0] = *(const uint4*)(brow + kc + kh);        // global_load_b128
        ub.q[1] = *(const uint4*)(brow + kc + 16 + kh);
        acc = __builtin_amdgcn_wmma_f32_16x16x32_bf16(false, ua.v, false, ub.v,
                                                      (short)0, acc, false, false);
      }
      int mo = (lane >> 4) * 8;
      #pragma unroll
      for (int v = 0; v < 8; ++v)
        ldsXW[(tm * 16 + mo + v) * GG + tn * 16 + m] = acc[v];
    }

    if (tid < UU) { ldsH[tid] = hS[l * UU + tid]; ldsC[tid] = cS[l * UU + tid]; }
    const float bsum = bi[tid] + bh[tid];

    // prefetch next layer's weights into cache during the long recurrence
    if (l + 1 < NB) {
      const bf16* nWih = WihR + (size_t)l * GG * UU;          // layer l+1 Wih
      const bf16* nWhh = WhhA + (size_t)(l + 1) * GG * UU;    // layer l+1 Whh
      for (int p = tid; p < 1024; p += 512) {                 // 128B granularity
        __builtin_prefetch(nWih + (size_t)p * 64, 0, 1);
        __builtin_prefetch(nWhh + (size_t)p * 64, 0, 1);
      }
    }
#if HAVE_TDM
    if (wave == 0) __builtin_amdgcn_s_wait_tensorcnt(0);      // Whh tile landed
#endif
    __syncthreads();  // ldsXW + ldsWhB ready

    // recurrence over 64 steps: gate j = tid reads its padded Whh row (b128)
    const uint4*  w4 = (const uint4*)(ldsWhB + (size_t)tid * ROW_BF16);
    const float4* h4 = (const float4*)ldsH;
    for (int t = 0; t < NB; ++t) {
      float dot = 0.f;
      #pragma unroll 4
      for (int q = 0; q < 16; ++q) {
        union { uint4 u; bf16 b[8]; } cw;
        cw.u = w4[q];                                   // ds_load_b128 (8 weights)
        float4 ha = h4[2 * q], hb2 = h4[2 * q + 1];     // ds_load_b128 broadcast
        dot = fmaf((float)cw.b[0], ha.x, dot);
        dot = fmaf((float)cw.b[1], ha.y, dot);
        dot = fmaf((float)cw.b[2], ha.z, dot);
        dot = fmaf((float)cw.b[3], ha.w, dot);
        dot = fmaf((float)cw.b[4], hb2.x, dot);
        dot = fmaf((float)cw.b[5], hb2.y, dot);
        dot = fmaf((float)cw.b[6], hb2.z, dot);
        dot = fmaf((float)cw.b[7], hb2.w, dot);
      }
      ldsG[tid] = ldsXW[t * GG + tid] + bsum + dot;
      __syncthreads();
      if (tid < UU) {
        float ig = sigmoidf_(ldsG[tid]);
        float fg = sigmoidf_(ldsG[UU + tid]);
        float g2 = tanhf(ldsG[2 * UU + tid]);
        float og = sigmoidf_(ldsG[3 * UU + tid]);
        float cn = fmaf(fg, ldsC[tid], ig * g2);
        float hn = og * tanhf(cn);
        ldsC[tid] = cn; ldsH[tid] = hn;
        if (l < NB - 1)      ldsA[t * UU + tid] = (bf16)hn;   // next layer's input
        else if (hsOut)      hsOut[t * UU + tid] = hn;        // last-layer sequence
      }
      __syncthreads();
    }
    if (tid < UU) { hS[l * UU + tid] = ldsH[tid]; cS[l * UU + tid] = ldsC[tid]; }
  }
}

// ---------------- faithful scrambling reshape ----------------
__global__ void scramble(float* __restrict__ out, const float* __restrict__ encOuts) {
  for (int i = blockIdx.x * blockDim.x + threadIdx.x; i < NB * LB * UU;
       i += gridDim.x * blockDim.x) {
    int b = i / (LB * UU), r = i % (LB * UU), t = r / UU, u = r % UU;
    int k = u * LB + t;
    out[i] = encOuts[((k / UU) * NB + b) * UU + (k % UU)];
  }
}

// ---------------- decoder attention: scores ----------------
__global__ void dec_attn1(const float* __restrict__ encOut, const float* __restrict__ hS,
                          const float* __restrict__ ddW, const float* __restrict__ ddB,
                          const float* __restrict__ dlW, const float* __restrict__ dlB,
                          float* __restrict__ escore) {
  int bt = blockIdx.x, b = bt / LB, t = bt % LB, u = threadIdx.x;   // block 128
  const float* w  = ddW + u * (2 * UU);
  const float* eo = encOut + (size_t)(b * LB + t) * UU;
  const float* hd = hS + b * UU;
  float acc = ddB[u];
  for (int k = 0; k < UU; ++k) acc = fmaf(w[k], eo[k], acc);
  for (int k = 0; k < UU; ++k) acc = fmaf(w[UU + k], hd[k], acc);
  __shared__ float red[UU];
  red[u] = dlW[u] * tanhf(acc);
  __syncthreads();
  for (int st = 64; st > 0; st >>= 1) {
    if (u < st) red[u] += red[u + st];
    __syncthreads();
  }
  if (u == 0) escore[b * LB + t] = red[0] + dlB[0];
}

// ---------------- decoder attention: softmax + context -> dec_in (bf16, K=256) ----------------
__global__ void dec_attn2(const float* __restrict__ escore, const float* __restrict__ encOut,
                          bf16* __restrict__ Abuf) {
  int b = blockIdx.x, u = threadIdx.x;              // block 128
  __shared__ float al[LB];
  if (u < LB) al[u] = escore[b * LB + u];
  __syncthreads();
  float mx = -1e30f;
  for (int t = 0; t < LB; ++t) mx = fmaxf(mx, al[t]);
  float se = 0.f;
  for (int t = 0; t < LB; ++t) se += expf(al[t] - mx);
  float ctx = 0.f;
  for (int t = 0; t < LB; ++t)
    ctx = fmaf(expf(al[t] - mx) / se, encOut[(size_t)(b * LB + t) * UU + u], ctx);
  Abuf[b * 2 * UU + u]      = (bf16)ctx;            // [ctx, zeros]
  Abuf[b * 2 * UU + UU + u] = (bf16)0.0f;
}

// ---------------- output head (bias-only fc, batch-constant) ----------------
__global__ void out_head(const float* __restrict__ fcB, const float* __restrict__ outW,
                         const float* __restrict__ outB, float* __restrict__ out, int i) {
  int j = threadIdx.x;                              // block 64
  __shared__ float red[64];
  __shared__ float sres;
  red[j] = outW[j] * tanhf(fcB[j]);
  __syncthreads();
  for (int st = 32; st > 0; st >>= 1) {
    if (j < st) red[j] += red[j + st];
    __syncthreads();
  }
  if (j == 0) sres = 1.0f / (1.0f + expf(-(red[0] + outB[0])));
  __syncthreads();
  out[j * NOUT + i] = sres;                         // every batch row gets the same value
}

// ---------------- host orchestration ----------------
extern "C" void kernel_launch(void* const* d_in, const int* in_sizes, int n_in,
                              void* d_out, int out_size, void* d_ws, size_t ws_size,
                              hipStream_t stream) {
  (void)in_sizes; (void)n_in; (void)out_size; (void)ws_size;
  const float* x        = (const float*)d_in[0];
  const float* encWih0  = (const float*)d_in[1];
  const float* encWhh0  = (const float*)d_in[2];
  const float* encBih0  = (const float*)d_in[3];
  const float* encBhh0  = (const float*)d_in[4];
  const float* encWih   = (const float*)d_in[5];
  const float* encWhh   = (const float*)d_in[6];
  const float* encBih   = (const float*)d_in[7];
  const float* encBhh   = (const float*)d_in[8];
  const float* attnW    = (const float*)d_in[9];
  const float* attnB    = (const float*)d_in[10];
  const float* decWih0  = (const float*)d_in[11];
  const float* decWhh0  = (const float*)d_in[12];
  const float* decBih0  = (const float*)d_in[13];
  const float* decBhh0  = (const float*)d_in[14];
  const float* decWih   = (const float*)d_in[15];
  const float* decWhh   = (const float*)d_in[16];
  const float* decBih   = (const float*)d_in[17];
  const float* decBhh   = (const float*)d_in[18];
  const float* ddW      = (const float*)d_in[19];
  const float* ddB      = (const float*)d_in[20];
  const float* dlW      = (const float*)d_in[21];
  const float* dlB      = (const float*)d_in[22];
  const float* fcB      = (const float*)d_in[24];
  const float* outW     = (const float*)d_in[25];
  const float* outB     = (const float*)d_in[26];
  float* outp = (float*)d_out;

  // workspace bump allocator (256B aligned)
  char* w = (char*)d_ws;
  auto alloc = [&](size_t bytes) { void* p = (void*)w; w += (bytes + 255) & ~(size_t)255; return p; };
  bf16* encWih0p = (bf16*)alloc((size_t)512 * 32 * 2);
  bf16* encWihR  = (bf16*)alloc((size_t)63 * 512 * 128 * 2);
  bf16* encWhhA  = (bf16*)alloc((size_t)64 * 512 * 128 * 2);
  bf16* decWih0b = (bf16*)alloc((size_t)512 * 256 * 2);
  bf16* decWihR  = (bf16*)alloc((size_t)63 * 512 * 128 * 2);
  bf16* decWhhA  = (bf16*)alloc((size_t)64 * 512 * 128 * 2);
  float* hS      = (float*)alloc((size_t)64 * 128 * 4);
  float* cS      = (float*)alloc((size_t)64 * 128 * 4);
  bf16* Abuf     = (bf16*)alloc((size_t)64 * 256 * 2);
  float* encOuts = (float*)alloc((size_t)LB * 64 * 128 * 4);
  float* encOut  = (float*)alloc((size_t)64 * LB * 128 * 4);
  float* escore  = (float*)alloc((size_t)64 * LB * 4);

  (void)hipFuncSetAttribute((const void*)multilayer_lstm,
                            hipFuncAttributeMaxDynamicSharedMemorySize, LDS_TOTAL);

  // weight conversion (bf16 weights; fp32 accumulate)
  pad_wih0<<<64, 256, 0, stream>>>(encWih0p, encWih0);
  cvt_f32_bf16<<<2048, 256, 0, stream>>>(encWihR, encWih, 63 * 512 * 128);
  cvt_f32_bf16<<<256, 256, 0, stream>>>(encWhhA, encWhh0, 512 * 128);
  cvt_f32_bf16<<<2048, 256, 0, stream>>>(encWhhA + 512 * 128, encWhh, 63 * 512 * 128);
  cvt_f32_bf16<<<512, 256, 0, stream>>>(decWih0b, decWih0, 512 * 256);
  cvt_f32_bf16<<<2048, 256, 0, stream>>>(decWihR, decWih, 63 * 512 * 128);
  cvt_f32_bf16<<<256, 256, 0, stream>>>(decWhhA, decWhh0, 512 * 128);
  cvt_f32_bf16<<<2048, 256, 0, stream>>>(decWhhA + 512 * 128, decWhh, 63 * 512 * 128);
  zero_f32<<<32, 256, 0, stream>>>(hS, 64 * 128);
  zero_f32<<<32, 256, 0, stream>>>(cS, 64 * 128);

  // ---- encoder: 50 sequential steps ----
  for (int s = 0; s < LB; ++s) {
    enc_attn<<<64, 32, 0, stream>>>(x, attnW, attnB, hS, Abuf, s);
    multilayer_lstm<<<1, 512, LDS_TOTAL, stream>>>(
        Abuf, 32, encWih0p, encWihR, encWhhA,
        encBih0, encBhh0, encBih, encBhh, hS, cS, encOuts + (size_t)s * 64 * 128);
  }
  scramble<<<1600, 256, 0, stream>>>(encOut, encOuts);

  // ---- decoder: fresh zero state, 3 sequential steps ----
  zero_f32<<<32, 256, 0, stream>>>(hS, 64 * 128);
  zero_f32<<<32, 256, 0, stream>>>(cS, 64 * 128);
  for (int i = 0; i < NOUT; ++i) {
    dec_attn1<<<64 * LB, 128, 0, stream>>>(encOut, hS,
        ddW + (size_t)i * 128 * 256, ddB + i * 128, dlW + i * 128, dlB + i, escore);
    dec_attn2<<<64, 128, 0, stream>>>(escore, encOut, Abuf);
    multilayer_lstm<<<1, 512, LDS_TOTAL, stream>>>(
        Abuf, 256, decWih0b, decWihR, decWhhA,
        decBih0, decBhh0, decBih, decBhh, hS, cS, nullptr);
    out_head<<<1, 64, 0, stream>>>(fcB + i * 64, outW + i * 64, outB + i, outp, i);
  }
}
